// TransformerDecoder_2576980378156
// MI455X (gfx1250) — matrix operand order repaired and checked by reading, not compiled, not run on hardware
//
#include <hip/hip_runtime.h>

// ---------------------------------------------------------------------------
// CDNA5 (gfx1250, wave32) ISAB Set-Transformer inference.
// Matrix path: v_wmma_f32_16x16x32_bf16 (fp32 accumulate), LN/softmax in fp32.
// ---------------------------------------------------------------------------

typedef __attribute__((ext_vector_type(16))) __bf16 v16bf;
typedef __attribute__((ext_vector_type(8)))  __bf16 v8bf;
typedef __attribute__((ext_vector_type(8)))  float  v8f;

#define NDST    16384
#define MAXDEG  32
#define DIM     128
#define DOUT    256
#define SB      136      // bf16 LDS row stride (pad 8)
#define SF      132      // f32  LDS row stride (pad 4)
#define LNEPS   1e-5f
#define SCALE   0.08838834764831843f   // 1/sqrt(128)

// ws layout (bytes)
#define WS_W    0                       // 14*32768 + 65536 = 524288 bf16 swizzled weights
#define WS_QP0  524288                  // 2 * 512 floats
#define WS_POOL 528384                  // 16384*128 bf16

__device__ __forceinline__ v8f wmma_bf(v16bf a, v16bf b, v8f c) {
  return __builtin_amdgcn_wmma_f32_16x16x32_bf16(false, a, false, b, (short)0, c, false, false);
}

// A fragment (16x32 bf16, MxK) from LDS row-major buffer, row stride SB.
// lane L: row m0+(L&15); elements 0..7 at K=k0+8*hi, 8..15 at K=k0+16+8*hi.
__device__ __forceinline__ v16bf load_a_frag(const __bf16* base, int m0, int k0, int lane) {
  const int hi = lane >> 4;
  const __bf16* p = base + (m0 + (lane & 15)) * SB + k0 + 8 * hi;
  v8bf c0 = *(const v8bf*)p;
  v8bf c1 = *(const v8bf*)(p + 16);
  return __builtin_shufflevector(c0, c1, 0,1,2,3,4,5,6,7,8,9,10,11,12,13,14,15);
}

// B fragment (32x16 bf16, KxN) from pre-swizzled global weights:
// frag (kt,nt) is 1KB; lane L holds 16 contiguous bf16 at lane*32 bytes.
__device__ __forceinline__ v16bf load_b_frag(const __bf16* w, int kt, int nt, int ntiles, int lane) {
  const __bf16* p = w + ((size_t)(kt * ntiles + nt) << 9) + (lane << 4);
  v8bf c0 = *(const v8bf*)p;
  v8bf c1 = *(const v8bf*)(p + 8);
  return __builtin_shufflevector(c0, c1, 0,1,2,3,4,5,6,7,8,9,10,11,12,13,14,15);
}

// GEMM: [16*MT,128] x [128,128] + bias, 4 waves split the 8 N-tiles (2 each).
// EPI: 0 -> store bf16 to dB ; 1 -> store f32 to dF ; 2 -> dF += relu(acc+bias)
template<int MT, int EPI>
__device__ void gemm128(const __bf16* sA, const __bf16* __restrict__ W,
                        const float* __restrict__ bias,
                        __bf16* dB, float* dF, int lane, int wv) {
  const int hi = lane >> 4, lr = lane & 15;
  const int nt0 = wv * 2, nt1 = nt0 + 1;
#pragma unroll
  for (int mt = 0; mt < MT; ++mt) {
    const int m0 = mt * 16;
    v8f a0 = {}, a1 = {};
#pragma unroll
    for (int k = 0; k < 4; ++k) {
      v16bf af = load_a_frag(sA, m0, k * 32, lane);
      a0 = wmma_bf(af, load_b_frag(W, k, nt0, 8, lane), a0);
      a1 = wmma_bf(af, load_b_frag(W, k, nt1, 8, lane), a1);
    }
    const float b0v = bias[nt0 * 16 + lr];
    const float b1v = bias[nt1 * 16 + lr];
#pragma unroll
    for (int r = 0; r < 8; ++r) {
      const int M = m0 + r + 8 * hi;
      float v0 = a0[r] + b0v, v1 = a1[r] + b1v;
      if (EPI == 0) {
        dB[M * SB + nt0 * 16 + lr] = (__bf16)v0;
        dB[M * SB + nt1 * 16 + lr] = (__bf16)v1;
      } else if (EPI == 1) {
        dF[M * SF + nt0 * 16 + lr] = v0;
        dF[M * SF + nt1 * 16 + lr] = v1;
      } else {
        dF[M * SF + nt0 * 16 + lr] += fmaxf(v0, 0.f);
        dF[M * SF + nt1 * 16 + lr] += fmaxf(v1, 0.f);
      }
    }
  }
}

// LayerNorm of one 128-wide row, full-wave cooperative; optional bf16 mirror.
__device__ void ln_row(float* row, const float* __restrict__ g, const float* __restrict__ bb,
                       __bf16* brow, int lane) {
  float x[4], s = 0.f, s2 = 0.f;
#pragma unroll
  for (int j = 0; j < 4; ++j) { x[j] = row[lane + 32 * j]; s += x[j]; s2 += x[j] * x[j]; }
#pragma unroll
  for (int o = 16; o > 0; o >>= 1) { s += __shfl_xor(s, o, 32); s2 += __shfl_xor(s2, o, 32); }
  const float mu = s * (1.f / 128.f);
  const float rs = rsqrtf(s2 * (1.f / 128.f) - mu * mu + LNEPS);
#pragma unroll
  for (int j = 0; j < 4; ++j) {
    const int f = lane + 32 * j;
    const float y = (x[j] - mu) * rs * g[f] + bb[f];
    row[f] = y;
    if (brow) brow[f] = (__bf16)y;
  }
}

struct MabB { const float *bq, *bk, *bv, *bo, *g0, *b0, *g1, *b1; };
struct LayerB { MabB m0, m1; };
struct AllB { LayerB L[2]; };

// ---------------------------------------------------------------------------
// Prep 1: swizzle 15 fp32 weight matrices into bf16 B-fragment layout.
// Matrix order: per layer {mab0.Wk, mab0.Wv, mab0.Wo, mab1.Wq, mab1.Wk,
// mab1.Wv, mab1.Wo} x2, then Wout [128,256].
// ---------------------------------------------------------------------------
struct SwzSrc { const float* s[15]; };

__global__ void __launch_bounds__(256) swz_kernel(SwzSrc S, __bf16* __restrict__ dst) {
  const int gid = blockIdx.x * blockDim.x + threadIdx.x;   // 0..262143
  int m, e, cols;
  if (gid < 14 * 16384) { m = gid >> 14; e = gid & 16383; cols = 128; }
  else                  { m = 14; e = gid - 14 * 16384; cols = 256; }
  const int frag = e >> 9, i = e & 511;
  const int ntiles = cols >> 4;
  const int kt = frag / ntiles, nt = frag % ntiles;
  const int ln = i >> 4, el = i & 15;
  const int K = kt * 32 + ((ln >> 4) << 3) + (el & 7) + ((el >> 3) << 4);
  const int N = nt * 16 + (ln & 15);
  dst[gid] = (__bf16)S.s[m][K * cols + N];
}

// Prep 2: batch-invariant inducing projection Qp0 = I @ Wq + bq per layer.
struct QpSrc { const float* I[2]; const float* Wq[2]; const float* bq[2]; };

__global__ void __launch_bounds__(128) qp0_kernel(QpSrc S, float* __restrict__ qp0) {
  const int l = blockIdx.x, n = threadIdx.x;
  for (int q = 0; q < 4; ++q) {
    float acc = S.bq[l][n];
    for (int d2 = 0; d2 < 128; ++d2) acc += S.I[l][q * 128 + d2] * S.Wq[l][d2 * 128 + n];
    qp0[l * 512 + q * 128 + n] = acc;
  }
}

// ---------------------------------------------------------------------------
// Main: one batch element per 4-wave workgroup; full 2-layer ISAB in LDS.
// ---------------------------------------------------------------------------
__global__ void __launch_bounds__(128) isab_kernel(
    const float* __restrict__ vfeat, const int* __restrict__ nbr,
    const int* __restrict__ deg, const __bf16* __restrict__ wsw,
    const float* __restrict__ qp0g, __bf16* __restrict__ pool, AllB P) {
  __shared__ __align__(16) __bf16 sV[32 * SB];   // activation v (bf16 A-source)
  __shared__ __align__(16) __bf16 sK[32 * SB];   // Kp
  __shared__ __align__(16) __bf16 sVp[32 * SB];  // Vp
  __shared__ __align__(16) __bf16 sH[16 * SB];   // padded H (rows 4..15 zero)
  __shared__ __align__(16) float  sO[32 * SF];   // f32 residual/LN buffer
  __shared__ float sQ0[512];                     // Qp0 of current layer
  __shared__ float sAt[4 * 4 * 33];              // mab0 attention probs

  const int tid = threadIdx.x, lane = tid & 31, wv = tid >> 5;
  const int b = blockIdx.x;
  const int degb = deg[b];

  // Gather neighbor features -> bf16 LDS (padded rows kept, masked later).
  for (int r = wv; r < 32; r += 4) {
    const int src = nbr[b * MAXDEG + r];
    const float4 vv = *(const float4*)(vfeat + (size_t)src * DIM + lane * 4);
    __bf16* d = sV + r * SB + lane * 4;
    d[0] = (__bf16)vv.x; d[1] = (__bf16)vv.y; d[2] = (__bf16)vv.z; d[3] = (__bf16)vv.w;
  }
  for (int i = tid; i < 12 * SB; i += 128) sH[4 * SB + i] = (__bf16)0.f;

#pragma unroll
  for (int l = 0; l < 2; ++l) {
    const __bf16* Wk0 = wsw + (size_t)(l * 7 + 0) * 16384;
    const __bf16* Wv0 = wsw + (size_t)(l * 7 + 1) * 16384;
    const __bf16* Wo0 = wsw + (size_t)(l * 7 + 2) * 16384;
    const __bf16* Wq1 = wsw + (size_t)(l * 7 + 3) * 16384;
    const __bf16* Wk1 = wsw + (size_t)(l * 7 + 4) * 16384;
    const __bf16* Wv1 = wsw + (size_t)(l * 7 + 5) * 16384;
    const __bf16* Wo1 = wsw + (size_t)(l * 7 + 6) * 16384;
    const MabB& M0 = P.L[l].m0;
    const MabB& M1 = P.L[l].m1;

    for (int i = tid; i < 512; i += 128) sQ0[i] = qp0g[l * 512 + i];
    __syncthreads();

    // ---- MAB0: inducing points attend to neighbors ----
    gemm128<2, 0>(sV, Wk0, M0.bk, sK,  nullptr, lane, wv);   // Kp [32,128]
    gemm128<2, 0>(sV, Wv0, M0.bv, sVp, nullptr, lane, wv);   // Vp [32,128]
    __syncthreads();

    // logits + masked softmax: wave = head h, lane = key k
    {
      const int h = wv, k = lane;
      float kh[32];
#pragma unroll
      for (int d2 = 0; d2 < 32; ++d2) kh[d2] = (float)sK[k * SB + h * 32 + d2];
      const bool valid = k < degb;
#pragma unroll
      for (int q = 0; q < 4; ++q) {
        float dv = 0.f;
#pragma unroll
        for (int d2 = 0; d2 < 32; ++d2) dv += sQ0[q * DIM + h * 32 + d2] * kh[d2];
        float lg = valid ? dv * SCALE : -1e9f;
        float mx = lg;
        for (int o = 16; o > 0; o >>= 1) mx = fmaxf(mx, __shfl_xor(mx, o, 32));
        const float e = __expf(lg - mx);
        float ss = e;
        for (int o = 16; o > 0; o >>= 1) ss += __shfl_xor(ss, o, 32);
        sAt[(h * 4 + q) * 33 + k] = e / ss;
      }
    }
    __syncthreads();
    // attention output + residual Qp0: wave = query q
    {
      const int q = wv;
#pragma unroll
      for (int j = 0; j < 4; ++j) {           // head j covers features [32j,32j+32)
        const int f = lane + 32 * j;
        float a = sQ0[q * DIM + f];
        const float* pr = sAt + (j * 4 + q) * 33;
        for (int k = 0; k < 32; ++k) a += pr[k] * (float)sVp[k * SB + f];
        sO[q * SF + f] = a;
      }
    }
    __syncthreads();
    ln_row(sO + wv * SF, M0.g0, M0.b0, sH + wv * SB, lane);  // LN0 rows 0..3 -> sH bf16
    __syncthreads();
    gemm128<1, 2>(sH, Wo0, M0.bo, nullptr, sO, lane, wv);    // O += relu(O@Wo+bo)
    __syncthreads();
    ln_row(sO + wv * SF, M0.g1, M0.b1, sH + wv * SB, lane);  // LN1 -> H bf16 in sH
    __syncthreads();

    // ---- MAB1: neighbors attend to inducing summary H ----
    gemm128<2, 1>(sV, Wq1, M1.bq, nullptr, sO, lane, wv);    // Qp f32 [32,128]
    gemm128<1, 0>(sH, Wk1, M1.bk, sK,  nullptr, lane, wv);   // Kp1 (rows 0..3 valid)
    gemm128<1, 0>(sH, Wv1, M1.bv, sVp, nullptr, lane, wv);   // Vp1
    __syncthreads();
    // attention: lane = query q, wave = head h (owns feature range [32h,32h+32))
    {
      const int q = lane, h = wv;
      float p[4], mx = -1e30f;
#pragma unroll
      for (int k = 0; k < 4; ++k) {
        float dv = 0.f;
#pragma unroll
        for (int d2 = 0; d2 < 32; ++d2)
          dv += sO[q * SF + h * 32 + d2] * (float)sK[k * SB + h * 32 + d2];
        p[k] = dv * SCALE;
        mx = fmaxf(mx, p[k]);
      }
      float ss = 0.f;
#pragma unroll
      for (int k = 0; k < 4; ++k) { p[k] = __expf(p[k] - mx); ss += p[k]; }
      const float inv = 1.f / ss;
      for (int df = 0; df < 32; ++df) {
        const int f = h * 32 + df;
        float a = 0.f;
#pragma unroll
        for (int k = 0; k < 4; ++k) a += p[k] * (float)sVp[k * SB + f];
        sO[q * SF + f] += a * inv;          // residual: O = Qp + attn
      }
    }
    __syncthreads();
    for (int i = 0; i < 8; ++i)             // LN0 all 32 rows -> sV bf16 (FF A-source)
      ln_row(sO + (wv * 8 + i) * SF, M1.g0, M1.b0, sV + (wv * 8 + i) * SB, lane);
    __syncthreads();
    gemm128<2, 2>(sV, Wo1, M1.bo, nullptr, sO, lane, wv);    // O += relu(O@Wo+bo)
    __syncthreads();
    for (int i = 0; i < 8; ++i)             // LN1 -> next-layer v (bf16 + f32)
      ln_row(sO + (wv * 8 + i) * SF, M1.g1, M1.b1, sV + (wv * 8 + i) * SB, lane);
    __syncthreads();
  }

  // Masked mean over true mailbox rows -> pooled bf16 row.
  {
    const int f = tid;                      // 128 threads == 128 features
    float s = 0.f;
    for (int r = 0; r < degb; ++r) s += sO[r * SF + f];
    pool[(size_t)b * DIM + f] = (__bf16)(s / (float)degb);
  }
}

// ---------------------------------------------------------------------------
// Final projection: out[16384,256] = pool[16384,128] @ Wout + bout (WMMA).
// One 16-row M-tile per wave; 16 N-tiles x 4 K-steps.
// ---------------------------------------------------------------------------
__global__ void __launch_bounds__(256) out_gemm(
    const __bf16* __restrict__ pool, const __bf16* __restrict__ woutsw,
    const float* __restrict__ bout, float* __restrict__ out) {
  const int lane = threadIdx.x & 31;
  const int gw = blockIdx.x * 8 + (threadIdx.x >> 5);
  const int m0 = gw * 16;
  const int hi = lane >> 4, lr = lane & 15;
  v16bf A[4];
#pragma unroll
  for (int k = 0; k < 4; ++k) {
    const __bf16* p = pool + (size_t)(m0 + lr) * DIM + k * 32 + 8 * hi;
    v8bf c0 = *(const v8bf*)p;
    v8bf c1 = *(const v8bf*)(p + 16);
    A[k] = __builtin_shufflevector(c0, c1, 0,1,2,3,4,5,6,7,8,9,10,11,12,13,14,15);
  }
#pragma unroll 1
  for (int nt = 0; nt < 16; ++nt) {
    v8f acc = {};
#pragma unroll
    for (int k = 0; k < 4; ++k)
      acc = wmma_bf(A[k], load_b_frag(woutsw, k, nt, 16, lane), acc);
    const float bv = bout[nt * 16 + lr];
#pragma unroll
    for (int r = 0; r < 8; ++r)
      out[(size_t)(m0 + r + 8 * hi) * DOUT + nt * 16 + lr] = acc[r] + bv;
  }
}

// ---------------------------------------------------------------------------
// Host launcher. Input flatten order (setup_inputs insertion order):
// 0 vfeat, 1 nbr_idx, 2 deg, 3 isab0.I, 4..15 isab0.mab0{Wq,bq,Wk,bk,Wv,bv,
// Wo,bo,g0,b0,g1,b1}, 16..27 isab0.mab1, 28 isab1.I, 29..40 isab1.mab0,
// 41..52 isab1.mab1, 53 Wout, 54 bout.
// ---------------------------------------------------------------------------
extern "C" void kernel_launch(void* const* d_in, const int* in_sizes, int n_in,
                              void* d_out, int out_size, void* d_ws, size_t ws_size,
                              hipStream_t stream) {
  (void)in_sizes; (void)n_in; (void)out_size; (void)ws_size;
  const float* vfeat = (const float*)d_in[0];
  const int*   nbr   = (const int*)d_in[1];
  const int*   deg   = (const int*)d_in[2];
  auto F = [&](int i) { return (const float*)d_in[i]; };

  __bf16* wsw  = (__bf16*)((char*)d_ws + WS_W);
  float*  qp0  = (float*)((char*)d_ws + WS_QP0);
  __bf16* pool = (__bf16*)((char*)d_ws + WS_POOL);

  const int mb0[2] = {4, 29}, mb1[2] = {16, 41}, Ii[2] = {3, 28};

  SwzSrc sw;
  for (int l = 0; l < 2; ++l) {
    sw.s[l * 7 + 0] = F(mb0[l] + 2);  // mab0.Wk
    sw.s[l * 7 + 1] = F(mb0[l] + 4);  // mab0.Wv
    sw.s[l * 7 + 2] = F(mb0[l] + 6);  // mab0.Wo
    sw.s[l * 7 + 3] = F(mb1[l] + 0);  // mab1.Wq
    sw.s[l * 7 + 4] = F(mb1[l] + 2);  // mab1.Wk
    sw.s[l * 7 + 5] = F(mb1[l] + 4);  // mab1.Wv
    sw.s[l * 7 + 6] = F(mb1[l] + 6);  // mab1.Wo
  }
  sw.s[14] = F(53);                   // Wout
  hipLaunchKernelGGL(swz_kernel, dim3(1024), dim3(256), 0, stream, sw, wsw);

  QpSrc qs;
  for (int l = 0; l < 2; ++l) { qs.I[l] = F(Ii[l]); qs.Wq[l] = F(mb0[l] + 0); qs.bq[l] = F(mb0[l] + 1); }
  hipLaunchKernelGGL(qp0_kernel, dim3(2), dim3(128), 0, stream, qs, qp0);

  AllB P;
  for (int l = 0; l < 2; ++l) {
    const int a = mb0[l], c = mb1[l];
    P.L[l].m0 = { F(a + 1), F(a + 3), F(a + 5), F(a + 7), F(a + 8), F(a + 9), F(a + 10), F(a + 11) };
    P.L[l].m1 = { F(c + 1), F(c + 3), F(c + 5), F(c + 7), F(c + 8), F(c + 9), F(c + 10), F(c + 11) };
  }
  hipLaunchKernelGGL(isab_kernel, dim3(NDST), dim3(128), 0, stream,
                     vfeat, nbr, deg, (const __bf16*)wsw, (const float*)qp0, pool, P);

  hipLaunchKernelGGL(out_gemm, dim3(128), dim3(256), 0, stream,
                     (const __bf16*)pool, (const __bf16*)(wsw + (size_t)14 * 16384),
                     F(54), (float*)d_out);
}
